// TaskMoE_42838003810423
// MI455X (gfx1250) — matrix operand order repaired and checked by reading, not compiled, not run on hardware
//
#include <hip/hip_runtime.h>
#include <hip/hip_bf16.h>
#include <stdint.h>

// ---------------- types ----------------
typedef __attribute__((ext_vector_type(16))) __bf16 v16bf;
typedef __attribute__((ext_vector_type(8)))  float  v8f;
typedef __attribute__((ext_vector_type(4)))  float  v4f;
typedef __attribute__((ext_vector_type(4)))  unsigned int v4u;

union FragA { v16bf v; unsigned int u[8]; };
union FragB { v16bf v; v4u q[2]; };

static __device__ __forceinline__ unsigned int pack2_bf16(float a, float b) {
    // round-to-nearest-even f32 -> bf16, packed (a in low 16, b in high 16)
    unsigned int ua = __float_as_uint(a); ua += 0x7FFFu + ((ua >> 16) & 1u);
    unsigned int ub = __float_as_uint(b); ub += 0x7FFFu + ((ub >> 16) & 1u);
    return (ua >> 16) | (ub & 0xFFFF0000u);
}

// Problem constants
#define Hdim 768
#define Edim 16
#define Tdim 8
#define Ldim 8192
#define Bdim 16
#define Ktop 4

// ---------------- kernel 1: gating (silu -> linear -> softmax -> top4) ----------------
__global__ __launch_bounds__(256)
void gating_kernel(const float* __restrict__ task_full,
                   const float* __restrict__ gate_w,
                   const float* __restrict__ gate_b,
                   float* __restrict__ out_active,   // [16]
                   float* __restrict__ out_pfq,      // [8*16] one-hot
                   float* __restrict__ ws_gates,     // [4]
                   int*   __restrict__ ws_idx) {     // [4]
    __shared__ float logits[Tdim][Edim];
    const int t = threadIdx.x;
    if (t < Tdim * Edim) {
        const int task = t >> 4, e = t & 15;
        float acc = gate_b[e];
        for (int h = 0; h < Hdim; ++h) {
            float v = task_full[task * Hdim + h];
            float s = v / (1.0f + __expf(-v));   // SiLU
            acc += s * gate_w[h * Edim + e];
        }
        logits[task][e] = acc;
    }
    __syncthreads();
    if (t < Tdim) {
        float m = logits[t][0];
        for (int e = 1; e < Edim; ++e) m = fmaxf(m, logits[t][e]);
        float p[Edim]; float s = 0.0f;
        for (int e = 0; e < Edim; ++e) { p[e] = __expf(logits[t][e] - m); s += p[e]; }
        const float inv = 1.0f / s;
        for (int e = 0; e < Edim; ++e) p[e] *= inv;
        bool used[Edim];
        for (int e = 0; e < Edim; ++e) used[e] = false;
        float g[Ktop]; int id[Ktop];
        for (int k = 0; k < Ktop; ++k) {
            int best = 0; float bv = -1.0f;
            for (int e = 0; e < Edim; ++e)
                if (!used[e] && p[e] > bv) { bv = p[e]; best = e; }
            used[best] = true; g[k] = bv; id[k] = best;
        }
        for (int e = 0; e < Edim; ++e) out_pfq[t * Edim + e] = used[e] ? 1.0f : 0.0f;
        if (t == 0) {
            for (int e = 0; e < Edim; ++e) out_active[e] = p[e];
            for (int k = 0; k < Ktop; ++k) { ws_gates[k] = g[k]; ws_idx[k] = id[k]; }
        }
    }
}

// ---------------- kernel 2: combine selected experts -> bf16 W_c, swizzled ----------------
// Logical W_c[k][n], k=input dim (H), n=output dim (H).
// Stored pre-swizzled in WMMA B-fragment order:
//   fragment f = kt*48 + nt  (kt: K/32, nt: N/16), 512 bf16 per fragment,
//   within fragment: lane = (kin>=16 ? 16 : 0) + nin; element = v*2 + (kk&1), v = (kk&15)>>1.
// GEMM then loads 16 contiguous bf16 per lane (fully coalesced 1KB per wave).
__global__ __launch_bounds__(256)
void combine_kernel(const float* __restrict__ expert_w,
                    const float* __restrict__ ws_gates,
                    const int*   __restrict__ ws_idx,
                    unsigned short* __restrict__ wc) {
    const int i = blockIdx.x * blockDim.x + threadIdx.x;
    if (i >= Hdim * Hdim) return;
    const float g0 = ws_gates[0], g1 = ws_gates[1], g2 = ws_gates[2], g3 = ws_gates[3];
    const size_t e0 = (size_t)ws_idx[0] * (Hdim * Hdim);
    const size_t e1 = (size_t)ws_idx[1] * (Hdim * Hdim);
    const size_t e2 = (size_t)ws_idx[2] * (Hdim * Hdim);
    const size_t e3 = (size_t)ws_idx[3] * (Hdim * Hdim);
    const float val = g0 * expert_w[e0 + i] + g1 * expert_w[e1 + i]
                    + g2 * expert_w[e2 + i] + g3 * expert_w[e3 + i];
    unsigned int b = __float_as_uint(val);
    b += 0x7FFFu + ((b >> 16) & 1u);
    const unsigned short h = (unsigned short)(b >> 16);

    const int k = i / Hdim, n = i % Hdim;
    const int kt = k >> 5, kin = k & 31, nt = n >> 4, nin = n & 15;
    const int lane = ((kin >> 4) << 4) | nin;
    const int kk = kin & 15;
    const size_t off = ((size_t)(kt * 48 + nt) * 32 + lane) * 16 + ((kk >> 1) * 2 + (kk & 1));
    wc[off] = h;
}

// ---------------- kernel 3: fill 1.0f for batches 1..15 ----------------
__global__ __launch_bounds__(256)
void fill_ones(v4f* __restrict__ dst, size_t n4) {
    size_t i = (size_t)blockIdx.x * blockDim.x + threadIdx.x;
    const size_t stride = (size_t)gridDim.x * blockDim.x;
    v4f one = {1.0f, 1.0f, 1.0f, 1.0f};
    for (; i < n4; i += stride) dst[i] = one;
}

// ---------------- kernel 4: out[0] = 1 + x0 @ W_c  (bf16 WMMA, f32 accumulate) ----------------
// Block = 256 threads (8 waves), wave tile 16(M)x64(N), block tile 32(M)x256(N).
// Grid: (768/256, 8192/32) = (3, 256). K loop: 24 x 32.
__global__ __launch_bounds__(256)
void moe_gemm(const float* __restrict__ x0,
              const unsigned short* __restrict__ wc,
              float* __restrict__ out) {
    const int lane  = threadIdx.x & 31;
    const int wave  = threadIdx.x >> 5;
    const int wM    = wave >> 2;              // 0..1
    const int wN    = wave & 3;               // 0..3
    const int mBase = blockIdx.y * 32 + wM * 16;
    const int ntBase = blockIdx.x * 16 + wN * 4;   // fragment-N index (n = nt*16)
    const int mRow  = mBase + (lane & 15);
    const int half  = (lane < 16) ? 0 : 8;
    const float* xrow = x0 + (size_t)mRow * Hdim;

    v8f acc[4] = {};

    for (int kt = 0; kt < 24; ++kt) {
        // A fragment: 16x32 bf16, per-lane conversion from f32 in contiguous k order (k = 16*h + s)
        FragA a;
        const v4f* p0 = (const v4f*)(xrow + kt * 32 + 2 * half);    // K = 2*half..2*half+7
        const v4f* p1 = (const v4f*)(xrow + kt * 32 + 2 * half + 8); // K = 2*half+8..2*half+15
        v4f r0 = p0[0], r1 = p0[1], r2 = p1[0], r3 = p1[1];
        a.u[0] = pack2_bf16(r0.x, r0.y);
        a.u[1] = pack2_bf16(r0.z, r0.w);
        a.u[2] = pack2_bf16(r1.x, r1.y);
        a.u[3] = pack2_bf16(r1.z, r1.w);
        a.u[4] = pack2_bf16(r2.x, r2.y);
        a.u[5] = pack2_bf16(r2.z, r2.w);
        a.u[6] = pack2_bf16(r3.x, r3.y);
        a.u[7] = pack2_bf16(r3.z, r3.w);

        #pragma unroll
        for (int j = 0; j < 4; ++j) {
            FragB b;
            const v4u* bp = (const v4u*)(wc +
                ((size_t)(kt * 48 + (ntBase + j)) * 32 + lane) * 16);
            b.q[0] = bp[0];
            b.q[1] = bp[1];
            acc[j] = __builtin_amdgcn_wmma_f32_16x16x32_bf16(
                         /*neg_a=*/false, a.v, /*neg_b=*/false, b.v,
                         /*c_mod=*/(short)0, acc[j],
                         /*reuse_a=*/false, /*reuse_b=*/false);
        }
    }

    // Store D: VGPR r -> M = r (lanes 0-15) / M = 8+r (lanes 16-31), N = lane%16
    const int mOff = mBase + ((lane < 16) ? 0 : 8);
    #pragma unroll
    for (int j = 0; j < 4; ++j) {
        const int n = (ntBase + j) * 16 + (lane & 15);
        #pragma unroll
        for (int r = 0; r < 8; ++r) {
            out[(size_t)(mOff + r) * Hdim + n] = 1.0f + acc[j][r];
        }
    }
}

// ---------------- launch ----------------
extern "C" void kernel_launch(void* const* d_in, const int* in_sizes, int n_in,
                              void* d_out, int out_size, void* d_ws, size_t ws_size,
                              hipStream_t stream) {
    const float* x         = (const float*)d_in[0];   // [16, 8192, 768]; batch 0 at base
    const float* task_full = (const float*)d_in[1];   // [8, 768]
    const float* gate_w    = (const float*)d_in[2];   // [768, 16]
    const float* gate_b    = (const float*)d_in[3];   // [16]
    const float* expert_w  = (const float*)d_in[4];   // [16, 768, 768]

    float* out = (float*)d_out;
    const size_t OUT_N = (size_t)Bdim * Ldim * Hdim;  // 100663296
    float* out_active = out + OUT_N;                  // [16]
    float* out_pfq    = out + OUT_N + Edim;           // [128]

    float* ws_gates   = (float*)d_ws;                               // 16 B
    int*   ws_idx     = (int*)((char*)d_ws + 16);                   // 16 B
    unsigned short* wc = (unsigned short*)((char*)d_ws + 64);       // 768*768*2 B = 1.18 MB

    // 1. gating (tiny)
    gating_kernel<<<1, 256, 0, stream>>>(task_full, gate_w, gate_b,
                                         out_active, out_pfq, ws_gates, ws_idx);

    // 2. combine 4 selected experts into one bf16 matrix (swizzled B-fragment layout)
    combine_kernel<<<(Hdim * Hdim + 255) / 256, 256, 0, stream>>>(expert_w, ws_gates, ws_idx, wc);

    // 3. batches 1..15 are exactly 1.0f (HBM-write bound: ~377 MB)
    fill_ones<<<4096, 256, 0, stream>>>((v4f*)(out + (size_t)Ldim * Hdim),
                                        (size_t)(Bdim - 1) * Ldim * Hdim / 4);

    // 4. batch 0: 1.0f + x0 @ W_c via v_wmma_f32_16x16x32_bf16 (L2-resident inputs)
    dim3 grid(Hdim / 256, Ldim / 32);  // (3, 256)
    moe_gemm<<<grid, 256, 0, stream>>>(x, wc, out);
}
